// MultiHeadAttention_36687610642849
// MI455X (gfx1250) — matrix operand order repaired and checked by reading, not compiled, not run on hardware
//
#include <hip/hip_runtime.h>

// ---------------------------------------------------------------------------
// MHA forward for MI455X (gfx1250, wave32).  All GEMMs via
// v_wmma_f32_16x16x32_f16 (f16 in, f32 accumulate).  Flash-style streaming
// softmax; K/V chunks staged into LDS with double-buffered
// global_load_async_to_lds_b128 (ASYNCcnt); mask pre-packed to bits.
// ---------------------------------------------------------------------------

#define B_  2
#define T_  2048
#define E_  1024
#define H_  16
#define HD_ 64
#define NEG_ (-1.0e9f)

typedef __attribute__((ext_vector_type(16))) _Float16 v16h;
typedef __attribute__((ext_vector_type(8)))  _Float16 v8h;
typedef __attribute__((ext_vector_type(4)))  _Float16 v4h;
typedef __attribute__((ext_vector_type(8)))  float    v8f;

__device__ __forceinline__ v16h cat8(v8h lo, v8h hi) {
    return __builtin_shufflevector(lo, hi, 0,1,2,3,4,5,6,7,8,9,10,11,12,13,14,15);
}

__device__ __forceinline__ v8f wmma_f16(v16h a, v16h b, v8f c) {
    // D = A(16x32,f16) x B(32x16,f16) + C(16x16,f32)
    return __builtin_amdgcn_wmma_f32_16x16x32_f16(
        /*neg_a=*/false, a, /*neg_b=*/false, b,
        /*c_mod=*/(short)0, c, /*reuse_a=*/false, /*reuse_b=*/false);
}

// Async 16B copy global->LDS, tracked by ASYNCcnt (ISA 08_async_tensor §4).
__device__ __forceinline__ void async_cp16(const void* lds_ptr, const void* gptr) {
    unsigned int  loff  = (unsigned int)(uintptr_t)lds_ptr;   // low 32b of flat = LDS offset
    unsigned long gaddr = (unsigned long)(uintptr_t)gptr;
    asm volatile("global_load_async_to_lds_b128 %0, %1, off"
                 :: "v"(loff), "v"(gaddr) : "memory");
}

// A-fragment (16x32, row-major source, stride in halves).
// Contiguous-k layout used here (same as B): lanes 0-15 row=lane, K=0..15 ; lanes 16-31 same rows,
// K=16..31.  Two contiguous 16B loads per lane.
__device__ __forceinline__ v16h load_afrag(const _Float16* base, int stride, int lane) {
    int r  = lane & 15;
    int k0 = (lane >> 4) << 4;          // 0 or 16
    const _Float16* p = base + (size_t)r * stride + k0;
    v8h lo = *(const v8h*)(p);
    v8h hi = *(const v8h*)(p + 8);
    return cat8(lo, hi);
}

// B-fragment (32x16): source holds rows of N (each row = 32 contiguous K
// halves).  Contiguous-k layout used here (same as A): lane holds column n=lane%16; lanes 0-15 K=0..15,
// lanes 16-31 K=16..31.  Two contiguous 16B loads per lane.
__device__ __forceinline__ v16h load_bfrag(const _Float16* base, int stride, int lane) {
    int n  = lane & 15;
    int kb = (lane >> 4) << 4;          // 0 or 16
    const _Float16* p = base + (size_t)n * stride + kb;
    v8h lo = *(const v8h*)(p);
    v8h hi = *(const v8h*)(p + 8);
    return cat8(lo, hi);
}

// ---------------------------------------------------------------------------
// fp32 -> f16 conversion (weights), vectorized x4
// ---------------------------------------------------------------------------
__global__ void cvt_f32_f16_x4(const float* __restrict__ s, _Float16* __restrict__ d, int n4) {
    int i = blockIdx.x * 256 + threadIdx.x;
    if (i < n4) {
        float4 f = ((const float4*)s)[i];
        v4h h = { (_Float16)f.x, (_Float16)f.y, (_Float16)f.z, (_Float16)f.w };
        ((v4h*)d)[i] = h;
    }
}

// ---------------------------------------------------------------------------
// Pack bool mask [B*T, T] (1 byte each) into bit words [B*T, T/32].
// ---------------------------------------------------------------------------
__global__ void pack_mask(const unsigned char* __restrict__ m,
                          unsigned int* __restrict__ bits, int nwords) {
    int i = blockIdx.x * 256 + threadIdx.x;
    if (i >= nwords) return;
    const uint4* p = (const uint4*)(m + (size_t)i * 32);
    uint4 a = p[0], b = p[1];
    unsigned int dw[8] = { a.x, a.y, a.z, a.w, b.x, b.y, b.z, b.w };
    unsigned int w = 0;
    #pragma unroll
    for (int j = 0; j < 32; ++j)
        w |= (((dw[j >> 2] >> ((j & 3) * 8)) & 0xffu) ? 1u : 0u) << j;
    bits[i] = w;
}

// ---------------------------------------------------------------------------
// Tiled GEMM: out = A[MxK] @ W[NxK]^T + bias
//   AF32   : 1 -> A is fp32 (converted to f16 while staging to LDS), 0 -> f16
//   OUTMODE: 0 -> f16 row-major ; 1 -> f16 transposed-V [B,H,HD,T] ; 2 -> f32
// WG = 256 threads (8 waves), tile 128(M) x 64(N), K-chunk 32.
// ---------------------------------------------------------------------------
template<int AF32, int OUTMODE>
__global__ __launch_bounds__(256) void gemm_wT(
    const void* __restrict__ Avoid, const _Float16* __restrict__ W,
    const float* __restrict__ bias,
    _Float16* __restrict__ out16, float* __restrict__ out32,
    int M, int N, int K)
{
    const int SA = 40;                         // padded LDS row stride (halves)
    __shared__ __align__(16) _Float16 lA[128 * 40];
    __shared__ __align__(16) _Float16 lB[64 * 40];

    const int tid  = threadIdx.x;
    const int lane = tid & 31;
    const int wave = tid >> 5;
    const int wm   = wave & 3;
    const int wn   = wave >> 2;
    const int Mbase = blockIdx.y * 128;
    const int Nbase = blockIdx.x * 64;

    v8f acc[2][2] = {};

    for (int kk = 0; kk < K; kk += 32) {
        if (AF32) {
            const float* A = (const float*)Avoid;
            int c4 = (tid & 7) * 4;
            int r0 = tid >> 3;
            #pragma unroll
            for (int rr = 0; rr < 4; ++rr) {
                int r = r0 + rr * 32;
                const float* gp = A + (size_t)(Mbase + r) * K + kk + c4;
                float4 f = *(const float4*)gp;
                __builtin_prefetch(gp + 32, 0, 1);
                _Float16* d = &lA[r * SA + c4];
                d[0] = (_Float16)f.x; d[1] = (_Float16)f.y;
                d[2] = (_Float16)f.z; d[3] = (_Float16)f.w;
            }
        } else {
            const _Float16* A = (const _Float16*)Avoid;
            int c8 = (tid & 3) * 8;
            int r0 = tid >> 2;
            #pragma unroll
            for (int rr = 0; rr < 2; ++rr) {
                int r = r0 + rr * 64;
                const _Float16* gp = A + (size_t)(Mbase + r) * K + kk + c8;
                v8h h = *(const v8h*)gp;
                __builtin_prefetch(gp + 32, 0, 1);
                *(v8h*)&lA[r * SA + c8] = h;
            }
        }
        {
            int c8 = (tid & 3) * 8;
            int r  = tid >> 2;
            const _Float16* gp = W + (size_t)(Nbase + r) * K + kk + c8;
            v8h h = *(const v8h*)gp;
            __builtin_prefetch(gp + 32, 0, 1);
            *(v8h*)&lB[r * SA + c8] = h;
        }
        __syncthreads();

        v16h a0 = load_afrag(&lA[(wm * 32 +  0) * SA], SA, lane);
        v16h a1 = load_afrag(&lA[(wm * 32 + 16) * SA], SA, lane);
        v16h b0 = load_bfrag(&lB[(wn * 32 +  0) * SA], SA, lane);
        v16h b1 = load_bfrag(&lB[(wn * 32 + 16) * SA], SA, lane);
        acc[0][0] = wmma_f16(a0, b0, acc[0][0]);
        acc[0][1] = wmma_f16(a0, b1, acc[0][1]);
        acc[1][0] = wmma_f16(a1, b0, acc[1][0]);
        acc[1][1] = wmma_f16(a1, b1, acc[1][1]);
        __syncthreads();
    }

    #pragma unroll
    for (int i = 0; i < 2; ++i) {
        #pragma unroll
        for (int j = 0; j < 2; ++j) {
            int gm0 = Mbase + wm * 32 + i * 16 + ((lane >> 4) << 3);
            int gn  = Nbase + wn * 32 + j * 16 + (lane & 15);
            float bsv = bias[gn];
            #pragma unroll
            for (int r = 0; r < 8; ++r) {
                int gm = gm0 + r;
                float val = acc[i][j][r] + bsv;
                if (OUTMODE == 0) {
                    out16[(size_t)gm * N + gn] = (_Float16)val;
                } else if (OUTMODE == 1) {
                    int b = gm >> 11, t = gm & (T_ - 1);
                    int h = gn >> 6,  hd = gn & (HD_ - 1);
                    out16[(((size_t)(b * H_ + h) * HD_ + hd) << 11) + t] = (_Float16)val;
                } else {
                    out32[(size_t)gm * N + gn] = val;
                }
            }
        }
    }
}

// ---------------------------------------------------------------------------
// Flash attention: 4 waves / block, one 16-query tile per wave, shared
// double-buffered async LDS staging of K (32x64) and V (64x32) chunks.
// qw/kw: f16 [B,T,E] ; vT: f16 [B,H,HD,T] ; mbits: u32 [B*T, T/32].
// ---------------------------------------------------------------------------
__global__ __launch_bounds__(128) void attn_kernel(
    const _Float16* __restrict__ qw, const _Float16* __restrict__ kw,
    const _Float16* __restrict__ vT, const unsigned int* __restrict__ mbits,
    _Float16* __restrict__ attn_out)
{
    const int SK = 72;                               // K row stride (halves)
    const int SV = 40;                               // V row stride (halves)
    __shared__ __align__(16) _Float16 lK[2][32 * SK];
    __shared__ __align__(16) _Float16 lV[2][64 * SV];
    __shared__ __align__(16) _Float16 lP[4][16 * 40];

    const int tid  = threadIdx.x;
    const int lane = tid & 31;
    const int wave = tid >> 5;
    const int b = blockIdx.z, h = blockIdx.y;
    const int q0 = (blockIdx.x * 4 + wave) * 16;
    const size_t bt = (size_t)b * T_;
    const _Float16* vb = vT + (((size_t)(b * H_ + h) * HD_) << 11);

    // Preload Q A-fragments (rows q0.., head h, two K-chunks covering HD=64)
    const _Float16* qbase = qw + (bt + q0) * E_ + h * HD_;
    v16h qa[2];
    {
        int r  = lane & 15;
        int k0 = (lane >> 4) << 4;
        #pragma unroll
        for (int c = 0; c < 2; ++c) {
            const _Float16* p = qbase + (size_t)r * E_ + c * 32 + k0;
            qa[c] = cat8(*(const v8h*)p, *(const v8h*)(p + 8));
        }
    }

    // Issue async staging of one 32-key chunk (4 async ops per wave).
    auto issue = [&](int buf, int kc) {
        #pragma unroll
        for (int i = 0; i < 2; ++i) {                // K: 32 keys x 128B
            int idx = tid + i * 128;                 // 0..255
            int key = idx >> 3, c = idx & 7;
            async_cp16(&lK[buf][key * SK + c * 8],
                       kw + (bt + kc + key) * E_ + h * HD_ + c * 8);
        }
        #pragma unroll
        for (int i = 0; i < 2; ++i) {                // V: 64 hd rows x 64B
            int idx = tid + i * 128;
            int hd = idx >> 2, c = idx & 3;
            async_cp16(&lV[buf][hd * SV + c * 8],
                       vb + (size_t)hd * T_ + kc + c * 8);
        }
    };

    float mrun[8], lrun[8];
    #pragma unroll
    for (int r = 0; r < 8; ++r) { mrun[r] = -1e30f; lrun[r] = 0.0f; }
    v8f o[4] = {};

    const float scale = 0.125f;                      // 1/sqrt(HD)
    _Float16* P = lP[wave];

    int buf = 0;
    issue(0, 0);
    for (int kc = 0; kc < T_; kc += 32) {
        const bool hasnext = (kc + 32) < T_;
        if (hasnext) issue(buf ^ 1, kc + 32);
        // In-order async completion: <=4 outstanding means current chunk done.
        if (hasnext) asm volatile("s_wait_asynccnt 0x4" ::: "memory");
        else         asm volatile("s_wait_asynccnt 0x0" ::: "memory");
        __syncthreads();

        // Hoist the mask word load so its latency hides under the WMMAs.
        unsigned int mw = mbits[(bt + q0 + (lane & 15)) * (T_ / 32) + (kc >> 5)];

        // ---- scores: S = Q . K^T for 32 keys (two 16-key tiles) ----
        // Issue ALL fragment loads first so DScnt waits can be partial.
        v16h kf[2][2];
        #pragma unroll
        for (int j = 0; j < 2; ++j) {
            #pragma unroll
            for (int c = 0; c < 2; ++c) {
                const _Float16* p = &lK[buf][(j * 16 + (lane & 15)) * SK
                                             + c * 32 + ((lane >> 4) << 4)];
                kf[j][c] = cat8(*(const v8h*)p, *(const v8h*)(p + 8));
            }
        }
        v8f s[2] = {};
        #pragma unroll
        for (int j = 0; j < 2; ++j) {
            #pragma unroll
            for (int c = 0; c < 2; ++c)
                s[j] = wmma_f16(qa[c], kf[j][c], s[j]);
        }

        // ---- mask (bit words) + online softmax ----
        const int n  = lane & 15;
        const int mh = (lane >> 4) << 3;
        #pragma unroll
        for (int r = 0; r < 8; ++r) {
            unsigned int wr = __shfl(mw, mh + r, 32);   // word of row mh+r
            float s0 = s[0][r] * scale;
            float s1 = s[1][r] * scale;
            if ((wr >> n) & 1u)        s0 = NEG_;
            if ((wr >> (16 + n)) & 1u) s1 = NEG_;
            float rmax = fmaxf(s0, s1);
            #pragma unroll
            for (int d = 1; d < 16; d <<= 1)
                rmax = fmaxf(rmax, __shfl_xor(rmax, d, 32));
            float mnew  = fmaxf(mrun[r], rmax);
            float alpha = __expf(mrun[r] - mnew);
            mrun[r] = mnew;
            float p0 = __expf(s0 - mnew);
            float p1 = __expf(s1 - mnew);
            float rs = p0 + p1;
            #pragma unroll
            for (int d = 1; d < 16; d <<= 1)
                rs += __shfl_xor(rs, d, 32);
            lrun[r] = lrun[r] * alpha + rs;
            #pragma unroll
            for (int t = 0; t < 4; ++t) o[t][r] *= alpha;
            int m = mh + r;
            P[m * 40 + n]      = (_Float16)p0;       // relayout for A-frag
            P[m * 40 + 16 + n] = (_Float16)p1;
        }

        // ---- O += P . V (4 hd-tiles).  LDS ops are wave-ordered. ----
        v16h pa = load_afrag(P, 40, lane);
        v16h vf[4];
        #pragma unroll
        for (int t = 0; t < 4; ++t) {
            const _Float16* p = &lV[buf][(t * 16 + (lane & 15)) * SV
                                         + ((lane >> 4) << 4)];
            vf[t] = cat8(*(const v8h*)p, *(const v8h*)(p + 8));
        }
        #pragma unroll
        for (int t = 0; t < 4; ++t)
            o[t] = wmma_f16(pa, vf[t], o[t]);

        __syncthreads();
        buf ^= 1;
    }

    // ---- epilogue: 1/l, ReLU, store f16 ----
    _Float16* ob = attn_out + (bt + q0) * E_ + h * HD_;
    const int nn = lane & 15;
    const int mh = (lane >> 4) << 3;
    #pragma unroll
    for (int r = 0; r < 8; ++r) {
        float inv = 1.0f / lrun[r];
        int m = mh + r;
        #pragma unroll
        for (int t = 0; t < 4; ++t) {
            float val = o[t][r] * inv;
            val = fmaxf(val, 0.0f);                  // ReLU
            ob[(size_t)m * E_ + t * 16 + nn] = (_Float16)val;
        }
    }
}

// ---------------------------------------------------------------------------
extern "C" void kernel_launch(void* const* d_in, const int* in_sizes, int n_in,
                              void* d_out, int out_size, void* d_ws, size_t ws_size,
                              hipStream_t stream) {
    (void)in_sizes; (void)n_in; (void)out_size; (void)ws_size;
    const float* q  = (const float*)d_in[0];
    const float* k  = (const float*)d_in[1];
    const float* v  = (const float*)d_in[2];
    const unsigned char* mask = (const unsigned char*)d_in[3];  // bool [B,T,T]
    const float* Wq = (const float*)d_in[4];
    const float* bq = (const float*)d_in[5];
    const float* Wk = (const float*)d_in[6];
    const float* bk = (const float*)d_in[7];
    const float* Wv = (const float*)d_in[8];
    const float* bv = (const float*)d_in[9];
    const float* Wo = (const float*)d_in[10];
    const float* bo = (const float*)d_in[11];
    float* out = (float*)d_out;

    const size_t EE  = (size_t)E_ * E_;          // 1M
    const size_t BTE = (size_t)B_ * T_ * E_;     // 4M
    _Float16* p     = (_Float16*)d_ws;           // ~41 MB total scratch
    _Float16* Wq16  = p;
    _Float16* Wk16  = p + EE;
    _Float16* Wv16  = p + 2 * EE;
    _Float16* Wo16  = p + 3 * EE;
    _Float16* qw16  = p + 4 * EE;
    _Float16* kw16  = qw16 + BTE;
    _Float16* vT16  = kw16 + BTE;
    _Float16* at16  = vT16 + BTE;
    unsigned int* mb = (unsigned int*)(at16 + BTE);

    // 1) weights -> f16 ; mask -> bits
    {
        int n4 = (int)(EE / 4);
        dim3 g((n4 + 255) / 256);
        cvt_f32_f16_x4<<<g, 256, 0, stream>>>(Wq, Wq16, n4);
        cvt_f32_f16_x4<<<g, 256, 0, stream>>>(Wk, Wk16, n4);
        cvt_f32_f16_x4<<<g, 256, 0, stream>>>(Wv, Wv16, n4);
        cvt_f32_f16_x4<<<g, 256, 0, stream>>>(Wo, Wo16, n4);
        int nwords = (int)((size_t)B_ * T_ * T_ / 32);
        pack_mask<<<dim3((nwords + 255) / 256), 256, 0, stream>>>(mask, mb, nwords);
    }

    const int M = B_ * T_;
    dim3 gg(E_ / 64, M / 128);

    // 2) projections (fp32 A -> f16 out; V stored transposed [B,H,HD,T])
    gemm_wT<1, 0><<<gg, 256, 0, stream>>>(q, Wq16, bq, qw16, nullptr, M, E_, E_);
    gemm_wT<1, 0><<<gg, 256, 0, stream>>>(k, Wk16, bk, kw16, nullptr, M, E_, E_);
    gemm_wT<1, 1><<<gg, 256, 0, stream>>>(v, Wv16, bv, vT16, nullptr, M, E_, E_);

    // 3) streaming-softmax attention + ReLU (async LDS staging)
    attn_kernel<<<dim3(T_ / 64, H_, B_), 128, 0, stream>>>(qw16, kw16, vT16, mb, at16);

    // 4) output projection (f16 A -> fp32 out + bias)
    gemm_wT<0, 2><<<gg, 256, 0, stream>>>(at16, Wo16, bo, nullptr, out, M, E_, E_);
}
